// CausalSelfAttention_26027501813824
// MI455X (gfx1250) — matrix operand order repaired and checked by reading, compiled, not run on hardware
//
#include <hip/hip_runtime.h>
#include <hip/hip_bf16.h>

#define N_HEAD   12
#define N_EMBD   768
#define BATCH    4
#define SEQ      2048
#define HEAD_DIM 64
#define BT       (BATCH * SEQ)       // 8192 rows
#define C3       (3 * N_EMBD)        // 2304

typedef __attribute__((ext_vector_type(16))) __bf16 v16bf;
typedef __attribute__((ext_vector_type(8)))  float  v8f;

union Frag16 {
  unsigned short u[16];
  uint4 q[2];
  v16bf v;
};
static_assert(sizeof(Frag16) == 32, "frag must be 8 VGPRs worth per lane pair");

__device__ __forceinline__ unsigned short f2bf(float f) {
  unsigned int u = __float_as_uint(f);
  return (unsigned short)((u + 0x7FFFu + ((u >> 16) & 1u)) >> 16);
}

#define WMMA_BF16(A, B, C) \
  __builtin_amdgcn_wmma_f32_16x16x32_bf16(false, (A), false, (B), (short)0, (C), false, false)

// Async copy of 16B/lane from global to LDS, tracked on ASYNCcnt.
__device__ __forceinline__ void async_b128_to_lds(const void* gptr, void* lptr) {
  unsigned lds = (unsigned)(unsigned long long)lptr;       // low 32b = LDS offset
  unsigned long long ga = (unsigned long long)gptr;
  asm volatile("global_load_async_to_lds_b128 %0, %1, off"
               :: "v"(lds), "v"(ga) : "memory");
}
__device__ __forceinline__ void wait_async0() {
  asm volatile("s_wait_asynccnt 0x0" ::: "memory");
}

// ---------------------------------------------------------------------------
// Elementwise f32 -> bf16 (coalesced both sides)
__global__ __launch_bounds__(256) void k_cvt_bf16(const float* __restrict__ src,
                                                  unsigned short* __restrict__ dst,
                                                  int n) {
  int i = blockIdx.x * 256 + threadIdx.x;
  if (i < n) dst[i] = f2bf(src[i]);
}

// f32 src[K][N] -> bf16 dst[N][K] via 32x32 LDS tile (coalesced both sides).
__global__ __launch_bounds__(256) void k_cvt_bf16_t(const float* __restrict__ src,
                                                    unsigned short* __restrict__ dst,
                                                    int K, int N) {
  __shared__ unsigned short tile[32][33];   // +1 pad: no bank conflicts
  const int tx = threadIdx.x & 31;
  const int ty = threadIdx.x >> 5;          // 0..7
  const int n0 = blockIdx.x * 32;
  const int k0 = blockIdx.y * 32;
#pragma unroll
  for (int i = 0; i < 4; ++i) {
    const int k = k0 + ty + i * 8;
    tile[ty + i * 8][tx] = f2bf(src[(size_t)k * N + n0 + tx]);
  }
  __syncthreads();
#pragma unroll
  for (int i = 0; i < 4; ++i) {
    const int n = n0 + ty + i * 8;
    dst[(size_t)n * K + k0 + tx] = tile[tx][ty + i * 8];
  }
}

// ---------------------------------------------------------------------------
// 2x2 register-blocked GEMM micro-kernel: one wave owns a 32x32 output block.
__device__ __forceinline__ void bf16_gemm_2x2(const unsigned short* __restrict__ a0,
                                              const unsigned short* __restrict__ a1,
                                              const unsigned short* __restrict__ b0,
                                              const unsigned short* __restrict__ b1,
                                              int Kdim, int half, v8f* acc) {
  for (int kc = 0; kc < Kdim; kc += 32) {
    if ((kc & 127) == 0) {   // one near-cache prefetch per 256B of advance
      __builtin_prefetch(a0 + kc + 512, 0, 3);
      __builtin_prefetch(a1 + kc + 512, 0, 3);
      __builtin_prefetch(b0 + kc + 512, 0, 3);
      __builtin_prefetch(b1 + kc + 512, 0, 3);
    }
    Frag16 fa0, fa1, fb0, fb1;
    fa0.q[0] = *(const uint4*)(a0 + kc + half * 16);
    fa0.q[1] = *(const uint4*)(a0 + kc + half * 16 + 8);
    fa1.q[0] = *(const uint4*)(a1 + kc + half * 16);
    fa1.q[1] = *(const uint4*)(a1 + kc + half * 16 + 8);
    fb0.q[0] = *(const uint4*)(b0 + kc + half * 16);
    fb0.q[1] = *(const uint4*)(b0 + kc + half * 16 + 8);
    fb1.q[0] = *(const uint4*)(b1 + kc + half * 16);
    fb1.q[1] = *(const uint4*)(b1 + kc + half * 16 + 8);
    acc[0] = WMMA_BF16(fa0.v, fb0.v, acc[0]);
    acc[1] = WMMA_BF16(fa0.v, fb1.v, acc[1]);
    acc[2] = WMMA_BF16(fa1.v, fb0.v, acc[2]);
    acc[3] = WMMA_BF16(fa1.v, fb1.v, acc[3]);
  }
}

// ---------------------------------------------------------------------------
// QKV projection: [8192x768] x [768x2304] + bias, scattered to
//   Q*0.125 -> [B,H,T,D] bf16 ; K -> [B,H,T,D] bf16 ; V -> [B,H,D,T] bf16 (transposed)
__global__ __launch_bounds__(256) void k_qkv(const unsigned short* __restrict__ xbf,
                                             const unsigned short* __restrict__ wTa,
                                             const float* __restrict__ b_attn,
                                             unsigned short* __restrict__ qbuf,
                                             unsigned short* __restrict__ kbuf,
                                             unsigned short* __restrict__ vT) {
  const int lane = threadIdx.x & 31, wave = threadIdx.x >> 5;
  const int half = lane >> 4, r = lane & 15;
  const int row0 = blockIdx.x * 32;
  const int col0 = (blockIdx.y * 8 + wave) * 32;

  v8f acc[4] = {{}, {}, {}, {}};
  bf16_gemm_2x2(xbf + (size_t)(row0 + r) * N_EMBD,
                xbf + (size_t)(row0 + 16 + r) * N_EMBD,
                wTa + (size_t)(col0 + r) * N_EMBD,
                wTa + (size_t)(col0 + 16 + r) * N_EMBD,
                N_EMBD, half, acc);

#pragma unroll
  for (int mi = 0; mi < 2; ++mi) {
#pragma unroll
    for (int ni = 0; ni < 2; ++ni) {
      const int col   = col0 + ni * 16 + r;
      const float bias = b_attn[col];
      const int which = col / N_EMBD;
      const int cc    = col % N_EMBD;
      const int head  = cc >> 6;
      const int d     = cc & 63;
      const int rowb  = row0 + mi * 16;
      const int bb    = rowb >> 11;
      const int tbase = rowb & (SEQ - 1);
      const size_t bh = (size_t)(bb * N_HEAD + head);
      const v8f a = acc[mi * 2 + ni];
#pragma unroll
      for (int v = 0; v < 8; ++v) {
        const int t = tbase + v + half * 8;
        const float val = a[v] + bias;
        if (which == 0) {
          qbuf[(bh * SEQ + t) * HEAD_DIM + d] = f2bf(val * 0.125f);  // fold 1/sqrt(64)
        } else if (which == 1) {
          kbuf[(bh * SEQ + t) * HEAD_DIM + d] = f2bf(val);
        } else {
          vT[(bh * HEAD_DIM + d) * SEQ + t] = f2bf(val);
        }
      }
    }
  }
}

// ---------------------------------------------------------------------------
// Flash attention, async-staged K/V.
// Block = 8 waves; wave w owns query rows [qTile*128 + w*16, +16).
// Per 32-key step: block cooperatively async-copies K(32x64) and V^T(64x32)
// bf16 tiles into LDS (double buffered, ASYNCcnt), one barrier, then each wave
// runs 4 WMMA (S) + online softmax + 4 WMMA (PV) off ds_load_b128 fragments.
__global__ __launch_bounds__(256) void k_attn(const unsigned short* __restrict__ qbuf,
                                              const unsigned short* __restrict__ kbuf,
                                              const unsigned short* __restrict__ vT,
                                              unsigned short* __restrict__ y) {
  __shared__ float pslab[8][16][32];              // wave-private P transpose slabs, 16KB
  __shared__ unsigned short kTile[2][32][64];     // [buf][key][d], 2 x 4KB
  __shared__ unsigned short vTile[2][64][32];     // [buf][d][key], 2 x 4KB

  const int tid  = threadIdx.x;
  const int lane = tid & 31, wave = tid >> 5;
  const int half = lane >> 4, r = lane & 15;
  const int bh     = blockIdx.y;                  // 0..47
  const int qTile0 = blockIdx.x * 128;
  const int qBase  = qTile0 + wave * 16;
  const int nIter  = (qTile0 + 128) / 32;         // uniform across block

  const unsigned short* qp = qbuf + (size_t)bh * SEQ * HEAD_DIM;
  const unsigned short* kp = kbuf + (size_t)bh * SEQ * HEAD_DIM;
  const unsigned short* vp = vT   + (size_t)bh * HEAD_DIM * SEQ;

  // Cooperative async stage of one 32-key step (block-uniform call sites only).
  auto stage = [&](int kb, int buf) {
    {  // K tile: 32 keys x 64 d, 128B/row -> 256 lanes x 16B
      const int key = tid >> 3, chunk = tid & 7;
      async_b128_to_lds(kp + (size_t)(kb + key) * HEAD_DIM + chunk * 8,
                        &kTile[buf][key][chunk * 8]);
    }
    {  // V tile: 64 d x 32 keys, 64B/row -> 256 lanes x 16B
      const int d = tid >> 2, chunk = tid & 3;
      async_b128_to_lds(vp + (size_t)d * SEQ + kb + chunk * 8,
                        &vTile[buf][d][chunk * 8]);
    }
  };

  // Q fragments for K-chunks d=[0,32) and [32,64)
  Frag16 qf[2];
#pragma unroll
  for (int c = 0; c < 2; ++c) {
    const unsigned short* qr = qp + (size_t)(qBase + r) * HEAD_DIM + c * 32 + half * 16;
    qf[c].q[0] = *(const uint4*)(qr);
    qf[c].q[1] = *(const uint4*)(qr + 8);
  }

  float m_i[8], l_i[8];
  v8f o0 = {}, o1 = {}, o2 = {}, o3 = {};
#pragma unroll
  for (int v = 0; v < 8; ++v) { m_i[v] = -1e30f; l_i[v] = 0.0f; }

  stage(0, 0);
  int kb = 0;
  for (int it = 0; it < nIter; ++it, kb += 32) {
    wait_async0();           // my staged tiles have landed
    __syncthreads();         // everyone's have landed; prev buffer fully consumed
    if (it + 1 < nIter) stage(kb + 32, (it + 1) & 1);   // overlap next stage
    const int buf = it & 1;

    // ---- S = Q * K^T for 32 keys (two 16-key C tiles), B-frags from LDS ----
    v8f s0 = {}, s1 = {};
#pragma unroll
    for (int c = 0; c < 2; ++c) {
      Frag16 b0, b1;
      const unsigned short* kr0 = &kTile[buf][r][c * 32 + half * 16];
      const unsigned short* kr1 = &kTile[buf][16 + r][c * 32 + half * 16];
      b0.q[0] = *(const uint4*)(kr0); b0.q[1] = *(const uint4*)(kr0 + 8);
      b1.q[0] = *(const uint4*)(kr1); b1.q[1] = *(const uint4*)(kr1 + 8);
      s0 = WMMA_BF16(qf[c].v, b0.v, s0);
      s1 = WMMA_BF16(qf[c].v, b1.v, s1);
    }

    // ---- causal mask + online softmax (row = v + half*8, col = lane&15) ----
    // Fully-masked steps are inert: alpha = exp(0) = 1, p = 0.
#pragma unroll
    for (int v = 0; v < 8; ++v) {
      const int qrow = qBase + v + half * 8;
      float a0 = (kb + r      <= qrow) ? s0[v] : -1e30f;
      float a1 = (kb + 16 + r <= qrow) ? s1[v] : -1e30f;
      float t = fmaxf(a0, a1);
#pragma unroll
      for (int mm = 1; mm < 16; mm <<= 1) t = fmaxf(t, __shfl_xor(t, mm));
      const float m_new = fmaxf(m_i[v], t);
      const float alpha = __expf(m_i[v] - m_new);
      const float p0 = __expf(a0 - m_new);
      const float p1 = __expf(a1 - m_new);
      float rs = p0 + p1;
#pragma unroll
      for (int mm = 1; mm < 16; mm <<= 1) rs += __shfl_xor(rs, mm);
      l_i[v] = l_i[v] * alpha + rs;
      m_i[v] = m_new;
      o0[v] *= alpha; o1[v] *= alpha; o2[v] *= alpha; o3[v] *= alpha;
      pslab[wave][v + half * 8][r]      = p0;
      pslab[wave][v + half * 8][16 + r] = p1;
    }

    // ---- P: C-layout -> A-fragment via wave-private LDS (no barrier needed) ----
    Frag16 pf;
    {
      const float* prow = &pslab[wave][r][0];
#pragma unroll
      for (int e = 0; e < 8; ++e) pf.u[e]     = f2bf(prow[half * 16 + e]);
#pragma unroll
      for (int e = 0; e < 8; ++e) pf.u[8 + e] = f2bf(prow[half * 16 + 8 + e]);
    }

    // ---- O += P * V, V^T B-frags from LDS: e -> key = half*16 + e ----
    {
      Frag16 vf;
      const unsigned short* vr;
      vr = &vTile[buf][0 * 16 + r][half * 16];
      vf.q[0] = *(const uint4*)(vr); vf.q[1] = *(const uint4*)(vr + 8);
      o0 = WMMA_BF16(pf.v, vf.v, o0);
      vr = &vTile[buf][1 * 16 + r][half * 16];
      vf.q[0] = *(const uint4*)(vr); vf.q[1] = *(const uint4*)(vr + 8);
      o1 = WMMA_BF16(pf.v, vf.v, o1);
      vr = &vTile[buf][2 * 16 + r][half * 16];
      vf.q[0] = *(const uint4*)(vr); vf.q[1] = *(const uint4*)(vr + 8);
      o2 = WMMA_BF16(pf.v, vf.v, o2);
      vr = &vTile[buf][3 * 16 + r][half * 16];
      vf.q[0] = *(const uint4*)(vr); vf.q[1] = *(const uint4*)(vr + 8);
      o3 = WMMA_BF16(pf.v, vf.v, o3);
    }
  }

  // ---- epilogue: y[b, qrow, h*64 + d] = O / l (bf16, merged-head layout) ----
  const int bb = bh / N_HEAD, hh = bh % N_HEAD;
#pragma unroll
  for (int v = 0; v < 8; ++v) {
    const int qrow = qBase + v + half * 8;
    const float inv = 1.0f / l_i[v];
    unsigned short* yr = y + ((size_t)(bb * SEQ + qrow)) * N_EMBD + hh * HEAD_DIM;
    yr[0 * 16 + r] = f2bf(o0[v] * inv);
    yr[1 * 16 + r] = f2bf(o1[v] * inv);
    yr[2 * 16 + r] = f2bf(o2[v] * inv);
    yr[3 * 16 + r] = f2bf(o3[v] * inv);
  }
}

// ---------------------------------------------------------------------------
// Output projection: [8192x768] x [768x768] + bias -> f32 out (2x2 blocked)
__global__ __launch_bounds__(256) void k_proj(const unsigned short* __restrict__ y,
                                              const unsigned short* __restrict__ wTp,
                                              const float* __restrict__ b_proj,
                                              float* __restrict__ out) {
  const int lane = threadIdx.x & 31, wave = threadIdx.x >> 5;
  const int half = lane >> 4, r = lane & 15;
  const int row0 = blockIdx.x * 32;
  const int col0 = (blockIdx.y * 8 + wave) * 32;

  v8f acc[4] = {{}, {}, {}, {}};
  bf16_gemm_2x2(y + (size_t)(row0 + r) * N_EMBD,
                y + (size_t)(row0 + 16 + r) * N_EMBD,
                wTp + (size_t)(col0 + r) * N_EMBD,
                wTp + (size_t)(col0 + 16 + r) * N_EMBD,
                N_EMBD, half, acc);

#pragma unroll
  for (int mi = 0; mi < 2; ++mi) {
#pragma unroll
    for (int ni = 0; ni < 2; ++ni) {
      const int col = col0 + ni * 16 + r;
      const float bias = b_proj[col];
      const v8f a = acc[mi * 2 + ni];
#pragma unroll
      for (int v = 0; v < 8; ++v)
        out[(size_t)(row0 + mi * 16 + v + half * 8) * N_EMBD + col] = a[v] + bias;
    }
  }
}

// ---------------------------------------------------------------------------
extern "C" void kernel_launch(void* const* d_in, const int* in_sizes, int n_in,
                              void* d_out, int out_size, void* d_ws, size_t ws_size,
                              hipStream_t stream) {
  const float* x      = (const float*)d_in[0];
  const float* w_attn = (const float*)d_in[1];
  const float* b_attn = (const float*)d_in[2];
  const float* w_proj = (const float*)d_in[3];
  const float* b_proj = (const float*)d_in[4];
  float* out = (float*)d_out;

  char* ws = (char*)d_ws;
  size_t off = 0;
  auto carve = [&](size_t bytes) {
    char* p = ws + off;
    off += (bytes + 255) & ~(size_t)255;
    return p;
  };
  unsigned short* xbf  = (unsigned short*)carve((size_t)BT * N_EMBD * 2);
  unsigned short* wTa  = (unsigned short*)carve((size_t)C3 * N_EMBD * 2);
  unsigned short* wTp  = (unsigned short*)carve((size_t)N_EMBD * N_EMBD * 2);
  unsigned short* qbuf = (unsigned short*)carve((size_t)BATCH * N_HEAD * SEQ * HEAD_DIM * 2);
  unsigned short* kbuf = (unsigned short*)carve((size_t)BATCH * N_HEAD * SEQ * HEAD_DIM * 2);
  unsigned short* vT   = (unsigned short*)carve((size_t)BATCH * N_HEAD * HEAD_DIM * SEQ * 2);
  unsigned short* ybf  = (unsigned short*)carve((size_t)BT * N_EMBD * 2);
  (void)ws_size; (void)in_sizes; (void)n_in; (void)out_size;

  {
    int n = BT * N_EMBD;
    k_cvt_bf16<<<(n + 255) / 256, 256, 0, stream>>>(x, xbf, n);
  }
  k_cvt_bf16_t<<<dim3(C3 / 32, N_EMBD / 32), 256, 0, stream>>>(w_attn, wTa, N_EMBD, C3);
  k_cvt_bf16_t<<<dim3(N_EMBD / 32, N_EMBD / 32), 256, 0, stream>>>(w_proj, wTp, N_EMBD, N_EMBD);

  k_qkv<<<dim3(BT / 32, C3 / 32 / 8), 256, 0, stream>>>(xbf, wTa, b_attn, qbuf, kbuf, vT);
  k_attn<<<dim3(SEQ / 128, BATCH * N_HEAD), 256, 0, stream>>>(qbuf, kbuf, vT, ybf);
  k_proj<<<dim3(BT / 32, N_EMBD / 32 / 8), 256, 0, stream>>>(ybf, wTp, b_proj, out);
}
